// CertViT_72748156060122
// MI455X (gfx1250) — matrix operand order repaired and checked by reading, not compiled, not run on hardware
//
#include <hip/hip_runtime.h>
#include <hip/hip_bf16.h>

// ---------------- model constants ----------------
#define BATCH   64
#define DIM     768
#define NHEAD   12
#define DHEAD   64
#define NTOK    197
#define NPATCH  196
#define DEPTH   12
#define SELL    3
#define MKEEP   137
#define NCLS    100
#define TROWS   (BATCH*NTOK)    // 12608
#define PROWS   (BATCH*NPATCH)  // 12544

typedef __attribute__((ext_vector_type(16))) _Float16 v16h;
typedef __attribute__((ext_vector_type(8)))  _Float16 v8h;
typedef __attribute__((ext_vector_type(8)))  float    v8f;

union F16x16 { v16h v; v8h h[2]; };

// ---------------- small helpers ----------------
__device__ __forceinline__ float gelu_f(float x) {
    float x3 = x * x * x;
    return 0.5f * x * (1.f + tanhf(0.7978845608028654f * (x + 0.044715f * x3)));
}

// CDNA5 async global->LDS copy (16B per active lane), tracked by ASYNCcnt.
__device__ __forceinline__ void async_copy_b128(unsigned lds_off, const void* gaddr) {
    asm volatile("global_load_async_to_lds_b128 %0, %1, off"
                 :: "v"(lds_off), "v"(gaddr) : "memory");
}
__device__ __forceinline__ void wait_asynccnt0() {
    asm volatile("s_wait_asynccnt 0x0" ::: "memory");
}

// ---------------- fp32 -> f16 convert (no transpose) ----------------
__global__ void k_cvt_f16(const float* __restrict__ src, _Float16* __restrict__ dst, int n) {
    int i = blockIdx.x * blockDim.x + threadIdx.x;
    int stride = gridDim.x * blockDim.x;
    for (; i < n; i += stride) dst[i] = (_Float16)src[i];
}

// ---------------- fp32 [L,K,N] -> f16 [L,N,K] convert + transpose ----------------
__global__ void k_cvt_f16_t(const float* __restrict__ src, _Float16* __restrict__ dst,
                            int L, int Kd, int Nd) {
    long total = (long)L * Kd * Nd;
    long i = (long)blockIdx.x * blockDim.x + threadIdx.x;
    long stride = (long)gridDim.x * blockDim.x;
    for (; i < total; i += stride) {
        long per = (long)Kd * Nd;
        long l = i / per;
        long rem = i - l * per;
        int n = (int)(rem / Kd);
        int k = (int)(rem - (long)n * Kd);
        dst[i] = (_Float16)src[l * per + (long)k * Nd + n];
    }
}

// ---------------- im2col: NCHW image -> [B*196, 768] f16 ----------------
__global__ void k_im2col(const float* __restrict__ in, _Float16* __restrict__ out) {
    int i = blockIdx.x * blockDim.x + threadIdx.x;
    const int total = PROWS * DIM;
    int stride = gridDim.x * blockDim.x;
    for (; i < total; i += stride) {
        int row = i / DIM, col = i - row * DIM;
        int b = row / NPATCH, p = row - b * NPATCH;
        int py = p / 14, px = p - py * 14;
        int c = col >> 8, rem = col & 255;
        int ph = rem >> 4, pw = rem & 15;
        out[i] = (_Float16)in[((b * 3 + c) * 224 + (py * 16 + ph)) * 224 + (px * 16 + pw)];
    }
}

// ---------------- assemble x = [cls ; patches] + pos ----------------
__global__ void k_assemble(const float* __restrict__ embed, const float* __restrict__ cls,
                           const float* __restrict__ pos, float* __restrict__ x) {
    int i = blockIdx.x * blockDim.x + threadIdx.x;
    const int total = TROWS * DIM;
    int stride = gridDim.x * blockDim.x;
    for (; i < total; i += stride) {
        int row = i / DIM, d = i - row * DIM;
        int b = row / NTOK, tok = row - b * NTOK;
        float v = (tok == 0) ? cls[d] : embed[(size_t)(b * NPATCH + tok - 1) * DIM + d];
        x[i] = v + pos[tok * DIM + d];
    }
}

// ---------------- layernorm (row-wise, D=768), f16 out ----------------
__global__ __launch_bounds__(256) void k_layernorm(
    const float* __restrict__ x, long rstride,
    const float* __restrict__ w, const float* __restrict__ b,
    _Float16* __restrict__ out) {
    __shared__ float s1[256], s2[256];
    int row = blockIdx.x, tid = threadIdx.x;
    const float* xr = x + (long)row * rstride;
    float a = 0.f, q = 0.f;
    for (int d = tid; d < DIM; d += 256) { float v = xr[d]; a += v; q += v * v; }
    s1[tid] = a; s2[tid] = q; __syncthreads();
    for (int s = 128; s > 0; s >>= 1) {
        if (tid < s) { s1[tid] += s1[tid + s]; s2[tid] += s2[tid + s]; }
        __syncthreads();
    }
    float mean = s1[0] * (1.f / DIM);
    float var  = s2[0] * (1.f / DIM) - mean * mean;
    float rs = rsqrtf(var + 1e-6f);
    for (int d = tid; d < DIM; d += 256)
        out[(long)row * DIM + d] = (_Float16)((xr[d] - mean) * rs * w[d] + b[d]);
}

// ---------------- token selection at layer SELL ----------------
// colsum_j = scale * (sum_i q_i) . k_j ; keep M smallest colsum (== M largest unc),
// ties broken toward lower index (matches jax top_k). Deterministic compaction.
__global__ __launch_bounds__(256) void k_select(const float* __restrict__ qkv,
                                                int* __restrict__ sel) {
    __shared__ float qs[DHEAD];
    __shared__ float cs[NTOK];
    __shared__ int   rk[NTOK];
    int bh = blockIdx.x;
    int h = bh % NHEAD, b = bh / NHEAD;
    int tid = threadIdx.x;
    const long base = (long)b * NTOK * (3 * DIM);
    if (tid < DHEAD) {
        float a = 0.f;
        for (int i2 = 0; i2 < NTOK; ++i2)
            a += qkv[base + (long)i2 * (3 * DIM) + h * DHEAD + tid];
        qs[tid] = a * 0.125f; // DH^-0.5
    }
    __syncthreads();
    if (tid < NTOK) {
        const float* kr = qkv + base + (long)tid * (3 * DIM) + DIM + h * DHEAD;
        float a = 0.f;
        #pragma unroll 8
        for (int d = 0; d < DHEAD; ++d) a += qs[d] * kr[d];
        cs[tid] = a;
    }
    __syncthreads();
    if (tid < NTOK) {
        float me = cs[tid]; int r = 0;
        for (int j = 0; j < NTOK; ++j) {
            float vj = cs[j];
            r += (vj < me) || (vj == me && j < tid);
        }
        rk[tid] = r;
    }
    __syncthreads();
    if (tid < NTOK && rk[tid] < MKEEP) {
        int pos = 0;
        for (int j = 0; j < tid; ++j) pos += (rk[j] < MKEEP);
        sel[bh * MKEEP + pos] = tid;
    }
}

// ---------------- attention: one block per (b, h, query) ----------------
__global__ __launch_bounds__(256) void k_attn(const float* __restrict__ qkv,
                                              const int* __restrict__ sel,
                                              _Float16* __restrict__ o, int Nk) {
    __shared__ float qs[DHEAD];
    __shared__ float sc[256];
    __shared__ int   id[256];
    __shared__ float red[256];
    int bid = blockIdx.x;
    int i = bid % NTOK; int bh = bid / NTOK;
    int h = bh % NHEAD, b = bh / NHEAD;
    int tid = threadIdx.x;
    const long base = (long)b * NTOK * (3 * DIM);
    const int hq = h * DHEAD, hk = DIM + h * DHEAD, hv = 2 * DIM + h * DHEAD;
    if (tid < Nk) id[tid] = sel ? sel[(b * NHEAD + h) * MKEEP + tid] : tid;
    if (tid < DHEAD) qs[tid] = qkv[base + (long)i * (3 * DIM) + hq + tid];
    __syncthreads();
    float s = -1e30f;
    if (tid < Nk) {
        const float* kr = qkv + base + (long)id[tid] * (3 * DIM) + hk;
        float acc = 0.f;
        #pragma unroll 8
        for (int d = 0; d < DHEAD; ++d) acc += qs[d] * kr[d];
        s = acc * 0.125f;
    }
    sc[tid] = s;
    red[tid] = s; __syncthreads();
    for (int st = 128; st > 0; st >>= 1) {
        if (tid < st) red[tid] = fmaxf(red[tid], red[tid + st]);
        __syncthreads();
    }
    float mx = red[0];
    __syncthreads();
    float p = (tid < Nk) ? __expf(sc[tid] - mx) : 0.f;
    sc[tid] = p;
    red[tid] = p; __syncthreads();
    for (int st = 128; st > 0; st >>= 1) {
        if (tid < st) red[tid] += red[tid + st];
        __syncthreads();
    }
    float inv = 1.f / red[0];
    __syncthreads();
    if (tid < DHEAD) {
        float acc = 0.f;
        for (int j = 0; j < Nk; ++j)
            acc += sc[j] * qkv[base + (long)id[j] * (3 * DIM) + hv + tid];
        o[(long)(b * NTOK + i) * DIM + h * DHEAD + tid] = (_Float16)(acc * inv);
    }
}

// ---------------- WMMA GEMM (NT): C = A(f16)[M,K] x Bt(f16)[N,K]^T ----------
// + bias (+res)(+gelu). Block tile 128x128, K-step 32, 8 waves (4x2),
// each wave 32x64 = 2x4 v_wmma_f32_16x16x32_f16 per K-step.
// Tiles staged with GLOBAL_LOAD_ASYNC_TO_LDS_B128 (ASYNCcnt), not VGPR round-trips.
__device__ __forceinline__ void store_frag(v8f acc, int rbase, int col, int Mr, int Nc,
                                           const float* bias, const float* res,
                                           float* outF, _Float16* outH, int gelu) {
    if (col >= Nc) return;
    float bv = bias ? bias[col] : 0.f;
    #pragma unroll
    for (int r = 0; r < 8; ++r) {
        int row = rbase + r;
        if (row >= Mr) break;
        float v = acc[r] + bv;
        if (gelu) v = gelu_f(v);
        if (res)  v += res[(long)row * Nc + col];
        if (outF) outF[(long)row * Nc + col] = v;
        if (outH) outH[(long)row * Nc + col] = (_Float16)v;
    }
}

__global__ __launch_bounds__(256) void k_gemm(
    const _Float16* __restrict__ A, const _Float16* __restrict__ Bt,
    const float* __restrict__ bias, const float* __restrict__ res,
    float* __restrict__ outF, _Float16* __restrict__ outH,
    int Mr, int Nc, int K, int gelu) {
    __shared__ __align__(16) _Float16 As[128 * 40];
    __shared__ __align__(16) _Float16 Bs[128 * 40];
    const int tid = threadIdx.x;
    const int lane = tid & 31, wave = tid >> 5;
    const int wm = wave >> 1, wn = wave & 1;   // 4 x 2 wave grid
    const int row0 = blockIdx.y * 128;
    const int col0 = blockIdx.x * 128;
    const int lm = lane & 15;
    const int akb = (lane < 16) ? 0 : 8;   // A frag: K base per half-wave
    const int bkb = (lane < 16) ? 0 : 8;   // B frag: K base per half-wave

    v8f zro = {};
    v8f acc[2][4];
    #pragma unroll
    for (int i = 0; i < 2; ++i)
        #pragma unroll
        for (int j = 0; j < 4; ++j) acc[i][j] = zro;

    for (int kt = 0; kt < K; kt += 32) {
        const int ktn = kt + 32;
        // ---- stage A tile [128 rows][32 halves] via async global->LDS ----
        #pragma unroll
        for (int c = tid; c < 512; c += 256) {
            int r = c >> 2, kc = (c & 3) << 3;
            int gr = row0 + r;
            if (gr < Mr) {
                async_copy_b128((unsigned)(size_t)&As[r * 40 + kc],
                                &A[(long)gr * K + kt + kc]);
                if (ktn < K) __builtin_prefetch(&A[(long)gr * K + ktn + kc], 0, 3);
            } else {
                #pragma unroll
                for (int q = 0; q < 8; ++q) As[r * 40 + kc + q] = (_Float16)0.f;
            }
        }
        // ---- stage B tile [128 n-rows][32 halves] from Bt[N,K] ----
        #pragma unroll
        for (int c = tid; c < 512; c += 256) {
            int n = c >> 2, kc = (c & 3) << 3;
            int gn = col0 + n;
            if (gn < Nc) {
                async_copy_b128((unsigned)(size_t)&Bs[n * 40 + kc],
                                &Bt[(long)gn * K + kt + kc]);
                if (ktn < K) __builtin_prefetch(&Bt[(long)gn * K + ktn + kc], 0, 3);
            } else {
                #pragma unroll
                for (int q = 0; q < 8; ++q) Bs[n * 40 + kc + q] = (_Float16)0.f;
            }
        }
        wait_asynccnt0();   // async LDS writes complete (ASYNCcnt==0)
        __syncthreads();

        // ---- fragments (ISA layouts): A lanes<16 K0-7/16-23, >=16 K8-15/24-31;
        //      B lane = N, same K order as A.
        F16x16 af[2], bf[4];
        #pragma unroll
        for (int i = 0; i < 2; ++i) {
            int ar = wm * 32 + i * 16 + lm;
            af[i].h[0] = *(const v8h*)&As[ar * 40 + akb];
            af[i].h[1] = *(const v8h*)&As[ar * 40 + akb + 16];
        }
        #pragma unroll
        for (int j = 0; j < 4; ++j) {
            int bn = wn * 64 + j * 16 + lm;
            bf[j].h[0] = *(const v8h*)&Bs[bn * 40 + bkb];
            bf[j].h[1] = *(const v8h*)&Bs[bn * 40 + bkb + 16];
        }
        #pragma unroll
        for (int j = 0; j < 4; ++j) {
            acc[0][j] = __builtin_amdgcn_wmma_f32_16x16x32_f16(false, af[0].v, false, bf[j].v, (short)0, acc[0][j], false, false);
            acc[1][j] = __builtin_amdgcn_wmma_f32_16x16x32_f16(false, af[1].v, false, bf[j].v, (short)0, acc[1][j], false, false);
        }
        __syncthreads();
    }

    // ---- epilogue: C/D layout VGPR r -> M=r (lanes 0-15) / M=r+8 (16-31), N=lane&15
    #pragma unroll
    for (int i = 0; i < 2; ++i) {
        int rbase = row0 + wm * 32 + i * 16 + ((lane >> 4) << 3);
        #pragma unroll
        for (int j = 0; j < 4; ++j) {
            int col = col0 + wn * 64 + j * 16 + lm;
            store_frag(acc[i][j], rbase, col, Mr, Nc, bias, res, outF, outH, gelu);
        }
    }
}

// ================= host-side orchestration =================
extern "C" void kernel_launch(void* const* d_in, const int* in_sizes, int n_in,
                              void* d_out, int out_size, void* d_ws, size_t ws_size,
                              hipStream_t stream) {
    (void)in_sizes; (void)n_in; (void)out_size; (void)ws_size;
    const float* inp     = (const float*)d_in[0];
    const float* patch_w = (const float*)d_in[1];
    const float* patch_b = (const float*)d_in[2];
    const float* cls_tok = (const float*)d_in[3];
    const float* pos_emb = (const float*)d_in[4];
    const float* qkv_w   = (const float*)d_in[5];
    const float* qkv_b   = (const float*)d_in[6];
    const float* proj_w  = (const float*)d_in[7];
    const float* proj_b  = (const float*)d_in[8];
    const float* n1w     = (const float*)d_in[9];
    const float* n1b     = (const float*)d_in[10];
    const float* n2w     = (const float*)d_in[11];
    const float* n2b     = (const float*)d_in[12];
    const float* fc1_w   = (const float*)d_in[13];
    const float* fc1_b   = (const float*)d_in[14];
    const float* fc2_w   = (const float*)d_in[15];
    const float* fc2_b   = (const float*)d_in[16];
    const float* nfw     = (const float*)d_in[17];
    const float* nfb     = (const float*)d_in[18];
    const float* head_w  = (const float*)d_in[19];
    const float* head_b  = (const float*)d_in[20];
    float* out = (float*)d_out;

    char* base = (char*)d_ws;
    size_t off = 0;
    auto carve = [&](size_t bytes) -> char* {
        char* p = base + off;
        off += (bytes + 255) & ~(size_t)255;
        return p;
    };
    // f16 weights (all stored [N,K] for NT GEMM)
    _Float16* wPatch = (_Float16*)carve((size_t)DIM * DIM * 2);            // already [N,K]
    _Float16* wQkvT  = (_Float16*)carve((size_t)DEPTH * DIM * 3 * DIM * 2);
    _Float16* wProjT = (_Float16*)carve((size_t)DEPTH * DIM * DIM * 2);
    _Float16* wFc1T  = (_Float16*)carve((size_t)DEPTH * DIM * 4 * DIM * 2);
    _Float16* wFc2T  = (_Float16*)carve((size_t)DEPTH * 4 * DIM * DIM * 2);
    _Float16* wHeadT = (_Float16*)carve((size_t)DIM * NCLS * 2);
    // activations
    _Float16* Xp    = (_Float16*)carve((size_t)PROWS * DIM * 2);
    float*    embed = (float*)   carve((size_t)PROWS * DIM * 4);
    float*    x     = (float*)   carve((size_t)TROWS * DIM * 4);
    _Float16* hbuf  = (_Float16*)carve((size_t)TROWS * DIM * 2);
    float*    qkvb  = (float*)   carve((size_t)TROWS * 3 * DIM * 4);
    _Float16* obuf  = (_Float16*)carve((size_t)TROWS * DIM * 2);
    _Float16* gbuf  = (_Float16*)carve((size_t)TROWS * 4 * DIM * 2);
    int*      selb  = (int*)     carve((size_t)BATCH * NHEAD * MKEEP * 4);
    _Float16* clsh  = (_Float16*)carve((size_t)BATCH * DIM * 2);

    auto cvt = [&](const float* s, _Float16* d, int n) {
        int blocks = (n + 255) / 256; if (blocks > 4096) blocks = 4096;
        k_cvt_f16<<<blocks, 256, 0, stream>>>(s, d, n);
    };
    auto cvtT = [&](const float* s, _Float16* d, int L, int Kd, int Nd) {
        k_cvt_f16_t<<<4096, 256, 0, stream>>>(s, d, L, Kd, Nd);
    };
    auto gemm = [&](const _Float16* Aa, const _Float16* Bb, const float* bi,
                    const float* rs2, float* oF, _Float16* oH,
                    int Mr, int Nc, int K, int gl) {
        dim3 g((Nc + 127) / 128, (Mr + 127) / 128);
        k_gemm<<<g, 256, 0, stream>>>(Aa, Bb, bi, rs2, oF, oH, Mr, Nc, K, gl);
    };

    // 1) weights -> f16, transposed to [N,K]
    cvt (patch_w, wPatch, DIM * DIM);                 // patch_w is already [D_out, K]
    cvtT(qkv_w,   wQkvT,  DEPTH, DIM,     3 * DIM);
    cvtT(proj_w,  wProjT, DEPTH, DIM,     DIM);
    cvtT(fc1_w,   wFc1T,  DEPTH, DIM,     4 * DIM);
    cvtT(fc2_w,   wFc2T,  DEPTH, 4 * DIM, DIM);
    cvtT(head_w,  wHeadT, 1,     DIM,     NCLS);

    // 2) patch embed: im2col + GEMM (+patch_b)
    k_im2col<<<4096, 256, 0, stream>>>(inp, Xp);
    gemm(Xp, wPatch, patch_b, nullptr, embed, nullptr, PROWS, DIM, DIM, 0);
    k_assemble<<<4096, 256, 0, stream>>>(embed, cls_tok, pos_emb, x);

    // 3) transformer layers
    for (int n = 0; n < DEPTH; ++n) {
        // LN1 -> h (f16)
        k_layernorm<<<TROWS, 256, 0, stream>>>(x, DIM, n1w + n * DIM, n1b + n * DIM, hbuf);
        // QKV = h @ qkv_w + b  (fp32 out)
        gemm(hbuf, wQkvT + (size_t)n * DIM * 3 * DIM, qkv_b + n * 3 * DIM, nullptr,
             qkvb, nullptr, TROWS, 3 * DIM, DIM, 0);
        // attention (key/value selection at layer SELL)
        if (n == SELL) {
            k_select<<<BATCH * NHEAD, 256, 0, stream>>>(qkvb, selb);
            k_attn<<<BATCH * NHEAD * NTOK, 256, 0, stream>>>(qkvb, selb, obuf, MKEEP);
        } else {
            k_attn<<<BATCH * NHEAD * NTOK, 256, 0, stream>>>(qkvb, nullptr, obuf, NTOK);
        }
        // x = x + o @ proj_w + proj_b   (residual fused)
        gemm(obuf, wProjT + (size_t)n * DIM * DIM, proj_b + n * DIM, x,
             x, nullptr, TROWS, DIM, DIM, 0);
        // LN2 -> h (f16)
        k_layernorm<<<TROWS, 256, 0, stream>>>(x, DIM, n2w + n * DIM, n2b + n * DIM, hbuf);
        // g = gelu(h @ fc1_w + b)  (f16 out)
        gemm(hbuf, wFc1T + (size_t)n * DIM * 4 * DIM, fc1_b + n * 4 * DIM, nullptr,
             nullptr, gbuf, TROWS, 4 * DIM, DIM, /*gelu=*/1);
        // x = x + g @ fc2_w + b   (residual fused)
        gemm(gbuf, wFc2T + (size_t)n * 4 * DIM * DIM, fc2_b + n * DIM, x,
             x, nullptr, TROWS, DIM, 4 * DIM, 0);
    }

    // 4) final LN on cls rows only, then head
    k_layernorm<<<BATCH, 256, 0, stream>>>(x, (long)NTOK * DIM, nfw, nfb, clsh);
    gemm(clsh, wHeadT, head_b, nullptr, out, nullptr, BATCH, NCLS, DIM, 0);
}
